// DAWN_35356170781342
// MI455X (gfx1250) — matrix operand order repaired and checked by reading, not compiled, not run on hardware
//
#include <hip/hip_runtime.h>

#define S_LEN 1024
#define DIMD  512
#define NHEAD 8
#define DHEAD 64
#define NCAND 144   // 9 cells * 16 block slots

typedef __attribute__((ext_vector_type(16))) _Float16 v16h;
typedef __attribute__((ext_vector_type(8)))  _Float16 v8h;
typedef __attribute__((ext_vector_type(8)))  float    v8f;

__device__ __forceinline__ v8f zero_v8f() {
  v8f z = {0.f,0.f,0.f,0.f,0.f,0.f,0.f,0.f};
  return z;
}

// DPP16 butterfly reduction within each 16-lane row (no LDS, pure VALU).
#define DPP_F32(x, ctrl) \
  __int_as_float(__builtin_amdgcn_update_dpp(0, __float_as_int(x), ctrl, 0xF, 0xF, true))

__device__ __forceinline__ float red_max16(float v) {
  v = fmaxf(v, DPP_F32(v, 0xB1));   // quad_perm [1,0,3,2]  (xor 1)
  v = fmaxf(v, DPP_F32(v, 0x4E));   // quad_perm [2,3,0,1]  (xor 2)
  v = fmaxf(v, DPP_F32(v, 0x141));  // row_half_mirror
  v = fmaxf(v, DPP_F32(v, 0x140));  // row_mirror
  return v;
}
__device__ __forceinline__ float red_add16(float v) {
  v += DPP_F32(v, 0xB1);
  v += DPP_F32(v, 0x4E);
  v += DPP_F32(v, 0x141);
  v += DPP_F32(v, 0x140);
  return v;
}

__device__ __forceinline__ v16h cat8(v8h lo, v8h hi) {
  return __builtin_shufflevector(lo, hi, 0,1,2,3,4,5,6,7,8,9,10,11,12,13,14,15);
}

// Async copy of one 16B chunk: global -> LDS, tracked by ASYNCcnt.
// VDST carries the LDS byte offset (low 32 bits of the generic LDS address, ISA 10.2).
__device__ __forceinline__ void async_g2l_b128(const void* gsrc, void* ldst) {
  const unsigned ldsoff = (unsigned)(uintptr_t)ldst;
  const unsigned long long ga = (unsigned long long)(uintptr_t)gsrc;
  asm volatile("global_load_async_to_lds_b128 %0, %1, off"
               :: "v"(ldsoff), "v"(ga) : "memory");
}
#define WAIT_ASYNC(n) asm volatile("s_wait_asynccnt %0" :: "i"(n) : "memory")

__device__ __forceinline__ float block_reduce_sum(float v, float* rbuf, int tid) {
  rbuf[tid] = v;
  __syncthreads();
  for (int s = 128; s > 0; s >>= 1) {
    if (tid < s) rbuf[tid] += rbuf[tid + s];
    __syncthreads();
  }
  float r = rbuf[0];
  __syncthreads();
  return r;
}

__device__ __forceinline__ int clip07(int v) { return v < 0 ? 0 : (v > 7 ? 7 : v); }

__device__ __forceinline__ int cell_coord(float p) {
  // jnp: clip(((pos - POS_MIN) / POS_RANGE * CELLS).astype(int32), 0, 7)
  return clip07((int)((p - (-3.0f)) / 6.0f * 8.0f));
}

// threshold_gate: per-candidate mixing weight (act * gate) into wout[0..143].
// Keep rule: kept iff (# strictly greater) < k  <=>  eg >= k-th largest (tie-keeping).
__device__ void gate_weights(const float* sv, float tauv, int k,
                             float* egb, float* wout, float* scal, int tid) {
  if (tid < NCAND) {
    float raw  = sv[tid] - tauv;
    float gate = raw > 0.f ? raw : 1e-8f * __expf(raw);
    egb[tid]   = __expf(gate) - 1.f;
  }
  __syncthreads();
  if (tid < NCAND) {
    float e = egb[tid];
    int cnt = 0;
    for (int j = 0; j < NCAND; ++j) cnt += (egb[j] > e) ? 1 : 0;
    wout[tid] = (cnt < k) ? e : 0.f;
  }
  __syncthreads();
  if (tid == 0) {
    float ssum = 0.f, smax = 0.f;
    for (int j = 0; j < NCAND; ++j) { ssum += wout[j]; smax = fmaxf(smax, wout[j]); }
    scal[0] = ssum; scal[1] = smax;
  }
  __syncthreads();
  if (tid < NCAND) {
    float g = wout[tid] / (scal[0] + 1e-8f) * tanhf(scal[1]);
    wout[tid] = g * sv[tid];   // act * gate (invalid -> g==0 -> weight 0)
  }
  __syncthreads();
}

// ---------------- Kernel 1: LN1 + routing positions + tau + cell candidate lists --------
__global__ __launch_bounds__(256)
void k1_ln_route(const float* __restrict__ x,
                 const float* __restrict__ ln1_s, const float* __restrict__ ln1_b,
                 const float* __restrict__ ppqk_k, const float* __restrict__ ppqk_b,
                 const float* __restrict__ ppv_k,  const float* __restrict__ ppv_b,
                 const float* __restrict__ ta_k,   const float* __restrict__ ta_b,
                 const int* __restrict__ qk_idx, const unsigned char* __restrict__ qk_valid,
                 const int* __restrict__ v_idx,  const unsigned char* __restrict__ v_valid,
                 float* __restrict__ xn_out, float* __restrict__ tau_out,
                 int* __restrict__ ci_qk, int* __restrict__ ci_v) {
  __shared__ float rbuf[256];
  __shared__ float res[7];
  const int s = blockIdx.x, tid = threadIdx.x;
  const float x0 = x[s * DIMD + tid];
  const float x1 = x[s * DIMD + tid + 256];
  const float mean = block_reduce_sum(x0 + x1, rbuf, tid) * (1.f / DIMD);
  const float d0 = x0 - mean, d1 = x1 - mean;
  const float var = block_reduce_sum(d0 * d0 + d1 * d1, rbuf, tid) * (1.f / DIMD);
  const float inv = rsqrtf(var + 1e-6f);
  const float n0 = d0 * inv * ln1_s[tid] + ln1_b[tid];
  const float n1 = d1 * inv * ln1_s[tid + 256] + ln1_b[tid + 256];
  xn_out[s * DIMD + tid] = n0;
  xn_out[s * DIMD + tid + 256] = n1;

  const float p0 = n0 * ppqk_k[tid * 2 + 0] + n1 * ppqk_k[(tid + 256) * 2 + 0];
  const float p1 = n0 * ppqk_k[tid * 2 + 1] + n1 * ppqk_k[(tid + 256) * 2 + 1];
  const float p2 = n0 * ppv_k[tid * 2 + 0]  + n1 * ppv_k[(tid + 256) * 2 + 0];
  const float p3 = n0 * ppv_k[tid * 2 + 1]  + n1 * ppv_k[(tid + 256) * 2 + 1];
  const float p4 = n0 * ta_k[tid * 3 + 0]   + n1 * ta_k[(tid + 256) * 3 + 0];
  const float p5 = n0 * ta_k[tid * 3 + 1]   + n1 * ta_k[(tid + 256) * 3 + 1];
  const float p6 = n0 * ta_k[tid * 3 + 2]   + n1 * ta_k[(tid + 256) * 3 + 2];
  float r;
  r = block_reduce_sum(p0, rbuf, tid); if (tid == 0) res[0] = r + ppqk_b[0];
  r = block_reduce_sum(p1, rbuf, tid); if (tid == 0) res[1] = r + ppqk_b[1];
  r = block_reduce_sum(p2, rbuf, tid); if (tid == 0) res[2] = r + ppv_b[0];
  r = block_reduce_sum(p3, rbuf, tid); if (tid == 0) res[3] = r + ppv_b[1];
  r = block_reduce_sum(p4, rbuf, tid); if (tid == 0) res[4] = r + ta_b[0];
  r = block_reduce_sum(p5, rbuf, tid); if (tid == 0) res[5] = r + ta_b[1];
  r = block_reduce_sum(p6, rbuf, tid); if (tid == 0) res[6] = r + ta_b[2];
  __syncthreads();

  if (tid < 3) tau_out[s * 4 + tid] = res[4 + tid];
  if (tid < NCAND) {
    const int o = tid / 16, b = tid % 16;
    const int ox = o / 3 - 1, oy = o % 3 - 1;
    {
      const int cx = cell_coord(res[0]), cy = cell_coord(res[1]);
      const int cell = clip07(cx + ox) * 8 + clip07(cy + oy);
      ci_qk[s * NCAND + tid] = qk_valid[cell * 16 + b] ? qk_idx[cell * 16 + b] : -1;
    }
    {
      const int cx = cell_coord(res[2]), cy = cell_coord(res[3]);
      const int cell = clip07(cx + ox) * 8 + clip07(cy + oy);
      ci_v[s * NCAND + tid] = v_valid[cell * 16 + b] ? v_idx[cell * 16 + b] : -1;
    }
  }
}

// ---------------- Kernel 2: candidate activations, top-k gates, Q/K/V mixtures ----------
__global__ __launch_bounds__(256)
void k2_qkv(const float* __restrict__ xn, const float* __restrict__ tau,
            const float* __restrict__ qk_neurons, const float* __restrict__ v_neurons,
            const int* __restrict__ ci_qk_g, const int* __restrict__ ci_v_g,
            _Float16* __restrict__ Qh, _Float16* __restrict__ KhT, _Float16* __restrict__ Vh) {
  __shared__ float sx[DIMD];
  __shared__ int ciq[NCAND], civ[NCAND];
  __shared__ float aq[NCAND], av[NCAND];
  __shared__ float egb[NCAND];
  __shared__ float wq[NCAND], wk[NCAND], wv[NCAND];
  __shared__ float scal[2];
  const int s = blockIdx.x, tid = threadIdx.x;
  sx[tid]       = xn[s * DIMD + tid];
  sx[tid + 256] = xn[s * DIMD + tid + 256];
  if (tid < NCAND) {
    ciq[tid] = ci_qk_g[s * NCAND + tid];
    civ[tid] = ci_v_g[s * NCAND + tid];
  }
  __syncthreads();

  const int wave = tid >> 5, lane = tid & 31;
  const int half = lane >> 4, sub = lane & 15;
  for (int base = wave * 2; base < NCAND; base += 16) {
    const int n = base + half;
    {
      const int idx = ciq[n];
      const float* row = qk_neurons + (size_t)(idx < 0 ? 0 : idx) * DIMD;
      const int nn = n + 16;
      if (nn < NCAND) {
        const int pidx = ciq[nn];
        __builtin_prefetch(qk_neurons + (size_t)(pidx < 0 ? 0 : pidx) * DIMD + sub * 32, 0, 1);
      }
      float p = 0.f;
      #pragma unroll
      for (int j = 0; j < 32; ++j) p += sx[sub + 16 * j] * row[sub + 16 * j];
      p = red_add16(p);
      if (sub == 0) aq[n] = (idx >= 0) ? p : -1e9f;
    }
    {
      const int idx = civ[n];
      const float* row = v_neurons + (size_t)(idx < 0 ? 0 : idx) * DIMD;
      const int nn = n + 16;
      if (nn < NCAND) {
        const int pidx = civ[nn];
        __builtin_prefetch(v_neurons + (size_t)(pidx < 0 ? 0 : pidx) * DIMD + sub * 32, 0, 1);
      }
      float p = 0.f;
      #pragma unroll
      for (int j = 0; j < 32; ++j) p += sx[sub + 16 * j] * row[sub + 16 * j];
      p = red_add16(p);
      if (sub == 0) av[n] = (idx >= 0) ? p : -1e9f;
    }
  }
  __syncthreads();

  gate_weights(aq, tau[s * 4 + 0], 32, egb, wq, scal, tid);
  gate_weights(aq, tau[s * 4 + 1], 32, egb, wk, scal, tid);
  gate_weights(av, tau[s * 4 + 2], 32, egb, wv, scal, tid);

  const int d0 = tid * 2;
  float q0 = 0.f, q1 = 0.f, k0 = 0.f, k1 = 0.f;
  for (int n = 0; n < NCAND; ++n) {
    const int idx = ciq[n];
    if (idx < 0) continue;
    const float w1 = wq[n], w2 = wk[n];
    if (w1 == 0.f && w2 == 0.f) continue;
    const float2 c = *(const float2*)(qk_neurons + (size_t)idx * DIMD + d0);
    q0 += w1 * c.x; q1 += w1 * c.y;
    k0 += w2 * c.x; k1 += w2 * c.y;
  }
  float v0 = 0.f, v1 = 0.f;
  for (int n = 0; n < NCAND; ++n) {
    const int idx = civ[n];
    if (idx < 0) continue;
    const float w = wv[n];
    if (w == 0.f) continue;
    const float2 c = *(const float2*)(v_neurons + (size_t)idx * DIMD + d0);
    v0 += w * c.x; v1 += w * c.y;
  }
  Qh[s * DIMD + d0]     = (_Float16)q0;
  Qh[s * DIMD + d0 + 1] = (_Float16)q1;
  KhT[(size_t)d0 * S_LEN + s]       = (_Float16)k0;   // K stored transposed [d][s]
  KhT[(size_t)(d0 + 1) * S_LEN + s] = (_Float16)k1;
  Vh[s * DIMD + d0]     = (_Float16)v0;
  Vh[s * DIMD + d0 + 1] = (_Float16)v1;
}

// ---------------- Kernel 3: causal flash attention, WMMA f16->f32 ------------------------
// One wave per (16-query tile, head). Online softmax. A: 16x32, B: 32x16, C/D: 16x16 f32.
// V tiles are double-buffered into LDS with global_load_async_to_lds_b128 (ASYNCcnt),
// issued one chunk ahead so L2 latency hides behind the score WMMAs + softmax.
__global__ __launch_bounds__(32)
void k3_attn(const _Float16* __restrict__ Qh, const _Float16* __restrict__ KhT,
             const _Float16* __restrict__ Vh, _Float16* __restrict__ aoh) {
  __shared__ __align__(16) _Float16 pTile[16 * 32];
  __shared__ __align__(16) _Float16 vbuf[2][32 * DHEAD];   // 2 x 4KB: 32 keys x 64 dh
  const int qt = blockIdx.x, head = blockIdx.y;
  const int lane = threadIdx.x;
  const int s0 = qt * 16;
  const int hb = head * DHEAD;
  const int mrow = lane & 15, grp = lane >> 4;
  const int kdB = (lane & 15) + 16 * grp;  // B-fragment contraction index for this lane

  // A fragments of Q (dh chunks k0=0 and k0=32): two contiguous 16B runs per lane.
  const _Float16* qrow = Qh + (size_t)(s0 + mrow) * DIMD + hb;
  const v16h aQ0 = cat8(*(const v8h*)(qrow + grp * 8),      *(const v8h*)(qrow + 16 + grp * 8));
  const v16h aQ1 = cat8(*(const v8h*)(qrow + 32 + grp * 8), *(const v8h*)(qrow + 48 + grp * 8));

  v8f acc[4];
  #pragma unroll
  for (int j = 0; j < 4; ++j) acc[j] = zero_v8f();
  float mi[8], li[8];
  #pragma unroll
  for (int v = 0; v < 8; ++v) { mi[v] = -1e30f; li[v] = 0.f; }

  const int nkeys = s0 + 16;  // causal: keys 0 .. s0+15
  const int nchunks = (nkeys + 31) / 32;

  // Issue async V-tile copy for a 32-key chunk: 4KB = 8 x b128 per lane.
  auto issue_v_tile = [&](int t0, _Float16* dst) {
    #pragma unroll
    for (int i = 0; i < 8; ++i) {
      const int t = lane + 32 * i;     // 256 x 16B chunks
      const int row = t >> 3;          // key within chunk (128B per key row)
      const int seg = t & 7;
      async_g2l_b128(Vh + (size_t)(t0 + row) * DIMD + hb + seg * 8,
                     dst + row * DHEAD + seg * 8);
    }
  };

  issue_v_tile(0, vbuf[0]);

  for (int ch = 0; ch < nchunks; ++ch) {
    const int t0 = ch * 32;
    const bool has_next = (ch + 1) < nchunks;
    if (has_next) issue_v_tile(t0 + 32, vbuf[(ch + 1) & 1]);

    float pv[2][8];
    float rowm[8];
    #pragma unroll
    for (int v = 0; v < 8; ++v) rowm[v] = -1e30f;

    #pragma unroll
    for (int c = 0; c < 2; ++c) {
      const int tc = t0 + c * 16;
      if (tc >= nkeys) {
        #pragma unroll
        for (int v = 0; v < 8; ++v) pv[c][v] = -1e30f;
        continue;
      }
      // B fragment of K^T from the transposed copy: lane = key, elements over dh (table k order).
      v16h bK0, bK1;
      #pragma unroll
      for (int i = 0; i < 16; ++i) {
        const int kd = (i < 8) ? (8 * grp + i) : (16 + 8 * grp + (i - 8));
        bK0[i] = KhT[(size_t)(hb + kd) * S_LEN + tc + (lane & 15)];
        bK1[i] = KhT[(size_t)(hb + 32 + kd) * S_LEN + tc + (lane & 15)];
      }
      v8f sc = zero_v8f();
      sc = __builtin_amdgcn_wmma_f32_16x16x32_f16(false, aQ0, false, bK0, (short)0, sc, false, false);
      sc = __builtin_amdgcn_wmma_f32_16x16x32_f16(false, aQ1, false, bK1, (short)0, sc, false, false);
      #pragma unroll
      for (int v = 0; v < 8; ++v) {
        const int row = v + 8 * grp;
        const int key = tc + (lane & 15);
        const float sv = (key <= s0 + row) ? sc[v] * 0.125f : -1e30f;  // 1/sqrt(64)
        pv[c][v] = sv;
        rowm[v] = fmaxf(rowm[v], sv);
      }
    }
    #pragma unroll
    for (int v = 0; v < 8; ++v) rowm[v] = red_max16(rowm[v]);

    float scalef[8];
    #pragma unroll
    for (int v = 0; v < 8; ++v) {
      const float nm = fmaxf(mi[v], rowm[v]);
      scalef[v] = __expf(mi[v] - nm);
      mi[v] = nm;
      li[v] *= scalef[v];
    }
    #pragma unroll
    for (int j = 0; j < 4; ++j)
      #pragma unroll
      for (int v = 0; v < 8; ++v) acc[j][v] *= scalef[v];

    float rs[8];
    #pragma unroll
    for (int v = 0; v < 8; ++v) rs[v] = 0.f;
    #pragma unroll
    for (int c = 0; c < 2; ++c)
      #pragma unroll
      for (int v = 0; v < 8; ++v) {
        const float p = __expf(pv[c][v] - mi[v]);
        rs[v] += p;
        pTile[(v + 8 * grp) * 32 + c * 16 + (lane & 15)] = (_Float16)p;
      }
    #pragma unroll
    for (int v = 0; v < 8; ++v) li[v] += red_add16(rs[v]);
    __syncthreads();

    // Drain: keep only the newest async batch (next chunk, 8/lane) outstanding;
    // current buffer is then guaranteed resident (async loads complete in order).
    if (has_next) { WAIT_ASYNC(8); } else { WAIT_ASYNC(0); }

    // P as A-fragment (16 x 32 keys): two contiguous 16B LDS reads per lane.
    const _Float16* prow = pTile + mrow * 32;
    const v16h aP = cat8(*(const v8h*)(prow + grp * 8), *(const v8h*)(prow + 16 + grp * 8));

    // V as B fragments from the LDS-staged tile: lane = dh column, elements over keys (table k order).
    const _Float16* vt = vbuf[ch & 1];
    #pragma unroll
    for (int j = 0; j < 4; ++j) {
      v16h bV;
      #pragma unroll
      for (int i = 0; i < 16; ++i) {
        const int kd = (i < 8) ? (8 * grp + i) : (16 + 8 * grp + (i - 8));
        bV[i] = vt[kd * DHEAD + j * 16 + (lane & 15)];
      }
      acc[j] = __builtin_amdgcn_wmma_f32_16x16x32_f16(false, aP, false, bV, (short)0, acc[j], false, false);
    }
    __syncthreads();
  }

  #pragma unroll
  for (int j = 0; j < 4; ++j)
    #pragma unroll
    for (int v = 0; v < 8; ++v) {
      const int row = v + 8 * grp, col = lane & 15;
      aoh[(size_t)(s0 + row) * DIMD + hb + j * 16 + col] = (_Float16)(acc[j][v] / li[v]);
    }
}

// ---------------- f32 -> f16 conversion --------------------------------------------------
__global__ __launch_bounds__(256)
void k_cvt_f16(const float* __restrict__ src, _Float16* __restrict__ dst, int n) {
  const int i = blockIdx.x * 256 + threadIdx.x;
  if (i < n) dst[i] = (_Float16)src[i];
}

// ---------------- Kernel 4: h = x + a_out @ expand_O (WMMA GEMM) -------------------------
__global__ __launch_bounds__(32)
void k4_proj(const _Float16* __restrict__ aoh, const _Float16* __restrict__ eOh,
             const float* __restrict__ x, float* __restrict__ hout) {
  const int n0 = blockIdx.x * 16, s0 = blockIdx.y * 16;
  const int lane = threadIdx.x;
  const int mrow = lane & 15, grp = lane >> 4;
  const int kdB = (lane & 15) + 16 * grp;
  v8f acc = zero_v8f();
  for (int k0 = 0; k0 < DIMD; k0 += 32) {
    const _Float16* arow = aoh + (size_t)(s0 + mrow) * DIMD + k0;
    const v16h a = cat8(*(const v8h*)(arow + grp * 8), *(const v8h*)(arow + 16 + grp * 8));
    v16h b;
    #pragma unroll
    for (int i = 0; i < 16; ++i) {
      const int kd = (i < 8) ? (8 * grp + i) : (16 + 8 * grp + (i - 8));
      b[i] = eOh[(size_t)(k0 + kd) * DIMD + n0 + (lane & 15)];
    }
    acc = __builtin_amdgcn_wmma_f32_16x16x32_f16(false, a, false, b, (short)0, acc, false, false);
  }
  #pragma unroll
  for (int v = 0; v < 8; ++v) {
    const int row = s0 + v + 8 * grp, col = n0 + (lane & 15);
    hout[(size_t)row * DIMD + col] = x[(size_t)row * DIMD + col] + acc[v];
  }
}

// ---------------- Kernel 5: knowledge circuit (LN2 + gather + gate k=64 + residual) ------
__global__ __launch_bounds__(256)
void k5_know(const float* __restrict__ h,
             const float* __restrict__ ln2_s, const float* __restrict__ ln2_b,
             const float* __restrict__ ppk_k, const float* __restrict__ ppk_b,
             const float* __restrict__ tk_k,  const float* __restrict__ tk_b,
             const int* __restrict__ know_idx, const unsigned char* __restrict__ know_valid,
             const float* __restrict__ know_neurons,
             float* __restrict__ out) {
  __shared__ float rbuf[256];
  __shared__ float shn[DIMD];
  __shared__ float res[3];
  __shared__ int ci[NCAND];
  __shared__ float ak[NCAND];
  __shared__ float egb[NCAND];
  __shared__ float wkn[NCAND];
  __shared__ float scal[2];
  const int s = blockIdx.x, tid = threadIdx.x;
  const float h0 = h[s * DIMD + tid];
  const float h1 = h[s * DIMD + tid + 256];
  const float mean = block_reduce_sum(h0 + h1, rbuf, tid) * (1.f / DIMD);
  const float d0 = h0 - mean, d1 = h1 - mean;
  const float var = block_reduce_sum(d0 * d0 + d1 * d1, rbuf, tid) * (1.f / DIMD);
  const float inv = rsqrtf(var + 1e-6f);
  const float n0 = d0 * inv * ln2_s[tid] + ln2_b[tid];
  const float n1 = d1 * inv * ln2_s[tid + 256] + ln2_b[tid + 256];
  shn[tid] = n0; shn[tid + 256] = n1;

  const float p0 = n0 * ppk_k[tid * 2 + 0] + n1 * ppk_k[(tid + 256) * 2 + 0];
  const float p1 = n0 * ppk_k[tid * 2 + 1] + n1 * ppk_k[(tid + 256) * 2 + 1];
  const float p2 = n0 * tk_k[tid] + n1 * tk_k[tid + 256];
  float r;
  r = block_reduce_sum(p0, rbuf, tid); if (tid == 0) res[0] = r + ppk_b[0];
  r = block_reduce_sum(p1, rbuf, tid); if (tid == 0) res[1] = r + ppk_b[1];
  r = block_reduce_sum(p2, rbuf, tid); if (tid == 0) res[2] = r + tk_b[0];
  __syncthreads();

  if (tid < NCAND) {
    const int o = tid / 16, b = tid % 16;
    const int ox = o / 3 - 1, oy = o % 3 - 1;
    const int cx = cell_coord(res[0]), cy = cell_coord(res[1]);
    const int cell = clip07(cx + ox) * 8 + clip07(cy + oy);
    ci[tid] = know_valid[cell * 16 + b] ? know_idx[cell * 16 + b] : -1;
  }
  __syncthreads();

  const int wave = tid >> 5, lane = tid & 31;
  const int half = lane >> 4, sub = lane & 15;
  for (int base = wave * 2; base < NCAND; base += 16) {
    const int n = base + half;
    const int idx = ci[n];
    const float* row = know_neurons + (size_t)(idx < 0 ? 0 : idx) * DIMD;
    const int nn = n + 16;
    if (nn < NCAND) {
      const int pidx = ci[nn];
      __builtin_prefetch(know_neurons + (size_t)(pidx < 0 ? 0 : pidx) * DIMD + sub * 32, 0, 1);
    }
    float p = 0.f;
    #pragma unroll
    for (int j = 0; j < 32; ++j) p += shn[sub + 16 * j] * row[sub + 16 * j];
    p = red_add16(p);
    if (sub == 0) ak[n] = (idx >= 0) ? p : -1e9f;
  }
  __syncthreads();

  gate_weights(ak, res[2], 64, egb, wkn, scal, tid);

  const int dd = tid * 2;
  float o0 = 0.f, o1 = 0.f;
  for (int n = 0; n < NCAND; ++n) {
    const int idx = ci[n];
    if (idx < 0) continue;
    const float w = wkn[n];
    if (w == 0.f) continue;
    const float2 c = *(const float2*)(know_neurons + (size_t)idx * DIMD + dd);
    o0 += w * c.x; o1 += w * c.y;
  }
  const float2 hv = *(const float2*)(h + (size_t)s * DIMD + dd);
  out[(size_t)s * DIMD + dd]     = hv.x + o0;
  out[(size_t)s * DIMD + dd + 1] = hv.y + o1;
}

// ---------------- Host launch ------------------------------------------------------------
extern "C" void kernel_launch(void* const* d_in, const int* in_sizes, int n_in,
                              void* d_out, int out_size, void* d_ws, size_t ws_size,
                              hipStream_t stream) {
  (void)in_sizes; (void)n_in; (void)out_size; (void)ws_size;
  const float* x            = (const float*)d_in[0];
  const float* qk_neurons   = (const float*)d_in[1];
  const float* v_neurons    = (const float*)d_in[2];
  const float* know_neurons = (const float*)d_in[3];
  const float* ppqk_k = (const float*)d_in[4];  const float* ppqk_b = (const float*)d_in[5];
  const float* ppv_k  = (const float*)d_in[6];  const float* ppv_b  = (const float*)d_in[7];
  const float* ppk_k  = (const float*)d_in[8];  const float* ppk_b  = (const float*)d_in[9];
  const float* ta_k   = (const float*)d_in[10]; const float* ta_b   = (const float*)d_in[11];
  const float* tk_k   = (const float*)d_in[12]; const float* tk_b   = (const float*)d_in[13];
  const float* expO   = (const float*)d_in[14];
  const float* ln1_s  = (const float*)d_in[15]; const float* ln1_b  = (const float*)d_in[16];
  const float* ln2_s  = (const float*)d_in[17]; const float* ln2_b  = (const float*)d_in[18];
  const int* qk_idx = (const int*)d_in[19];
  const unsigned char* qk_valid = (const unsigned char*)d_in[20];
  const int* v_idx = (const int*)d_in[21];
  const unsigned char* v_valid = (const unsigned char*)d_in[22];
  const int* know_idx = (const int*)d_in[23];
  const unsigned char* know_valid = (const unsigned char*)d_in[24];
  float* out = (float*)d_out;

  char* w = (char*)d_ws;
  auto alloc = [&](size_t bytes) -> char* {
    char* p = w;
    w += (bytes + 255) & ~(size_t)255;
    return p;
  };
  float*     xn    = (float*)alloc((size_t)S_LEN * DIMD * sizeof(float));
  float*     tau   = (float*)alloc((size_t)S_LEN * 4 * sizeof(float));
  int*       ci_qk = (int*)alloc((size_t)S_LEN * NCAND * sizeof(int));
  int*       ci_v  = (int*)alloc((size_t)S_LEN * NCAND * sizeof(int));
  _Float16*  Qh    = (_Float16*)alloc((size_t)S_LEN * DIMD * sizeof(_Float16));
  _Float16*  KhT   = (_Float16*)alloc((size_t)DIMD * S_LEN * sizeof(_Float16));
  _Float16*  Vh    = (_Float16*)alloc((size_t)S_LEN * DIMD * sizeof(_Float16));
  _Float16*  aoh   = (_Float16*)alloc((size_t)S_LEN * DIMD * sizeof(_Float16));
  _Float16*  eOh   = (_Float16*)alloc((size_t)DIMD * DIMD * sizeof(_Float16));
  float*     hbuf  = (float*)alloc((size_t)S_LEN * DIMD * sizeof(float));

  k1_ln_route<<<S_LEN, 256, 0, stream>>>(x, ln1_s, ln1_b, ppqk_k, ppqk_b, ppv_k, ppv_b,
                                         ta_k, ta_b, qk_idx, qk_valid, v_idx, v_valid,
                                         xn, tau, ci_qk, ci_v);
  k2_qkv<<<S_LEN, 256, 0, stream>>>(xn, tau, qk_neurons, v_neurons, ci_qk, ci_v, Qh, KhT, Vh);
  k3_attn<<<dim3(S_LEN / 16, NHEAD), 32, 0, stream>>>(Qh, KhT, Vh, aoh);
  k_cvt_f16<<<(DIMD * DIMD) / 256, 256, 0, stream>>>(expO, eOh, DIMD * DIMD);
  k4_proj<<<dim3(DIMD / 16, S_LEN / 16), 32, 0, stream>>>(aoh, eOh, x, hbuf);
  k5_know<<<S_LEN, 256, 0, stream>>>(hbuf, ln2_s, ln2_b, ppk_k, ppk_b, tk_k, tk_b,
                                     know_idx, know_valid, know_neurons, out);
}
